// TemporalGraphNetwork_74491912781913
// MI455X (gfx1250) — compile-verified, matrix operand order checked by reading, not run on hardware
//
#include <hip/hip_runtime.h>
#include <hip/hip_bf16.h>

#define N_NODES 100000
#define N_EDGES 500000
#define DM 128
#define HID 128
#define XDIM 320   // 2*DM + 64

typedef __attribute__((ext_vector_type(16))) __bf16 v16bf;
typedef __attribute__((ext_vector_type(8)))  float  v8f;
typedef __attribute__((ext_vector_type(4)))  unsigned int u32x4;

union BF16Frag { v16bf v; u32x4 q[2]; };

__device__ __forceinline__ unsigned short f32_to_bf16(float f) {
    unsigned int u = __float_as_uint(f);
    u += 0x7FFFu + ((u >> 16) & 1u);   // round-to-nearest-even
    return (unsigned short)(u >> 16);
}

__device__ __forceinline__ v8f wmma_bf16(const BF16Frag& a, const BF16Frag& b, v8f c) {
    return __builtin_amdgcn_wmma_f32_16x16x32_bf16(false, a.v, false, b.v, (short)0, c, false, false);
}

__device__ __forceinline__ float sigmf(float x) { return 1.0f / (1.0f + __expf(-x)); }

// ---- workspace layout (bf16 element offsets) ----
#define OFF_W1   0u        // 128x320
#define OFF_W2   40960u    // 128x128
#define OFF_WIH  57344u    // 384x128
#define OFF_WHH  106496u   // 384x128
#define OFF_WE   155648u   // 128x256
#define TOT_BF   188416u
#define OFF_WINNER_BYTES (TOT_BF * 2u)  // 376832, 4B aligned

// ------------------------------------------------------------------
__global__ __launch_bounds__(256) void tgn_convert_weights(
    const float* __restrict__ W1, const float* __restrict__ W2,
    const float* __restrict__ Wih, const float* __restrict__ Whh,
    const float* __restrict__ We, unsigned short* __restrict__ dst)
{
    unsigned int i = blockIdx.x * 256u + threadIdx.x;
    if (i >= TOT_BF) return;
    float v;
    if (i < OFF_W2)       v = W1[i];
    else if (i < OFF_WIH) v = W2[i - OFF_W2];
    else if (i < OFF_WHH) v = Wih[i - OFF_WIH];
    else if (i < OFF_WE)  v = Whh[i - OFF_WHH];
    else                  v = We[i - OFF_WE];
    dst[i] = f32_to_bf16(v);
}

__global__ __launch_bounds__(256) void tgn_init(
    const float* __restrict__ memory, float* __restrict__ updMem,
    int* __restrict__ winner)
{
    unsigned int i = blockIdx.x * 256u + threadIdx.x;
    if (i < (unsigned)N_NODES * DM) updMem[i] = memory[i];
    if (i < (unsigned)N_NODES) winner[i] = -1;
}

__global__ __launch_bounds__(256) void tgn_winner(
    const int* __restrict__ eidx, int* __restrict__ winner)
{
    unsigned int e = blockIdx.x * 256u + threadIdx.x;
    if (e < (unsigned)N_EDGES) atomicMax(&winner[eidx[e]], (int)e);
}

// ------------------------------------------------------------------
// Per-block: 16 edges, 8 waves; each wave owns a 16-wide N-tile.
__global__ __launch_bounds__(256) void tgn_edge_kernel(
    const int* __restrict__ eidx, const float* __restrict__ efeat,
    const float* __restrict__ memory,
    const unsigned short* __restrict__ W1bf, const float* __restrict__ b1,
    const unsigned short* __restrict__ W2bf, const float* __restrict__ b2,
    const unsigned short* __restrict__ Wihbf, const float* __restrict__ bih,
    const unsigned short* __restrict__ Whhbf, const float* __restrict__ bhh,
    const int* __restrict__ winner, float* __restrict__ outMem)
{
    __shared__ __attribute__((aligned(16))) unsigned short xA[16][XDIM];  // bf16 [mem_r|mem_c|ef]
    __shared__ float hF[16][DM];                                          // f32 h = mem[row]
    __shared__ __attribute__((aligned(16))) unsigned short h1A[16][HID];
    __shared__ __attribute__((aligned(16))) unsigned short msgA[16][HID];
    __shared__ int rowS[16], colS[16], winS[16];

    const int tid = threadIdx.x;
    const int e0  = blockIdx.x * 16;

    if (tid < 16) {
        int e = e0 + tid;
        int r = eidx[e];
        rowS[tid] = r;
        colS[tid] = eidx[N_EDGES + e];
        winS[tid] = (winner[r] == e) ? 1 : 0;
    }
    __syncthreads();

    // gather + f32->bf16 stage
    for (int idx = tid; idx < 16 * XDIM; idx += 256) {
        int m = idx / XDIM, k = idx - m * XDIM;
        float v;
        if (k < DM)          { v = memory[(size_t)rowS[m] * DM + k]; hF[m][k] = v; }
        else if (k < 2 * DM) { v = memory[(size_t)colS[m] * DM + (k - DM)]; }
        else                 { v = efeat[(size_t)(e0 + m) * 64 + (k - 2 * DM)]; }
        xA[m][k] = f32_to_bf16(v);
    }
    __syncthreads();

    const int wv   = tid >> 5;
    const int lane = tid & 31;
    const int hl   = lane >> 4;        // lane half
    const int mr   = lane & 15;        // A-row / B-col within tile
    const int koff = hl * 8;
    const int ncol = wv * 16 + mr;     // owned column in 128-wide outputs

    BF16Frag a, b;

    // ---- layer 1: h1 = relu(x @ W1^T + b1) ----
    v8f acc = {0.f,0.f,0.f,0.f,0.f,0.f,0.f,0.f};
#pragma unroll
    for (int k0 = 0; k0 < XDIM; k0 += 32) {
        a.q[0] = *(const u32x4*)&xA[mr][k0 + koff];
        a.q[1] = *(const u32x4*)&xA[mr][k0 + 16 + koff];
        const unsigned short* bp = W1bf + (size_t)ncol * XDIM + k0 + koff;
        b.q[0] = *(const u32x4*)bp;
        b.q[1] = *(const u32x4*)(bp + 16);
        acc = wmma_bf16(a, b, acc);
    }
    {
        float bias = b1[ncol];
#pragma unroll
        for (int r = 0; r < 8; ++r) {
            float v = acc[r] + bias;
            h1A[hl * 8 + r][ncol] = f32_to_bf16(v > 0.f ? v : 0.f);
        }
    }
    __syncthreads();

    // ---- layer 2: msg = h1 @ W2^T + b2 ----
    v8f acc2 = {0.f,0.f,0.f,0.f,0.f,0.f,0.f,0.f};
#pragma unroll
    for (int k0 = 0; k0 < HID; k0 += 32) {
        a.q[0] = *(const u32x4*)&h1A[mr][k0 + koff];
        a.q[1] = *(const u32x4*)&h1A[mr][k0 + 16 + koff];
        const unsigned short* bp = W2bf + (size_t)ncol * HID + k0 + koff;
        b.q[0] = *(const u32x4*)bp;
        b.q[1] = *(const u32x4*)(bp + 16);
        acc2 = wmma_bf16(a, b, acc2);
    }
    {
        float bias = b2[ncol];
#pragma unroll
        for (int r = 0; r < 8; ++r)
            msgA[hl * 8 + r][ncol] = f32_to_bf16(acc2[r] + bias);
    }
    __syncthreads();

    // ---- GRU gates: gi = msg @ W_ih^T, gh = h @ W_hh^T; r/z/n tiles share lane layout ----
    v8f gir = {0.f,0.f,0.f,0.f,0.f,0.f,0.f,0.f}, giz = gir, gin = gir;
    v8f ghr = gir, ghz = gir, ghn = gir;
#pragma unroll
    for (int k0 = 0; k0 < HID; k0 += 32) {
        a.q[0] = *(const u32x4*)&msgA[mr][k0 + koff];
        a.q[1] = *(const u32x4*)&msgA[mr][k0 + 16 + koff];
        const unsigned short* pr = Wihbf + (size_t)(0 * DM + ncol) * HID + k0 + koff;
        const unsigned short* pz = Wihbf + (size_t)(1 * DM + ncol) * HID + k0 + koff;
        const unsigned short* pn = Wihbf + (size_t)(2 * DM + ncol) * HID + k0 + koff;
        b.q[0] = *(const u32x4*)pr; b.q[1] = *(const u32x4*)(pr + 16);
        gir = wmma_bf16(a, b, gir);
        b.q[0] = *(const u32x4*)pz; b.q[1] = *(const u32x4*)(pz + 16);
        giz = wmma_bf16(a, b, giz);
        b.q[0] = *(const u32x4*)pn; b.q[1] = *(const u32x4*)(pn + 16);
        gin = wmma_bf16(a, b, gin);
    }
#pragma unroll
    for (int k0 = 0; k0 < HID; k0 += 32) {
        a.q[0] = *(const u32x4*)&xA[mr][k0 + koff];      // h (bf16) = first 128 cols of x
        a.q[1] = *(const u32x4*)&xA[mr][k0 + 16 + koff];
        const unsigned short* pr = Whhbf + (size_t)(0 * DM + ncol) * DM + k0 + koff;
        const unsigned short* pz = Whhbf + (size_t)(1 * DM + ncol) * DM + k0 + koff;
        const unsigned short* pn = Whhbf + (size_t)(2 * DM + ncol) * DM + k0 + koff;
        b.q[0] = *(const u32x4*)pr; b.q[1] = *(const u32x4*)(pr + 16);
        ghr = wmma_bf16(a, b, ghr);
        b.q[0] = *(const u32x4*)pz; b.q[1] = *(const u32x4*)(pz + 16);
        ghz = wmma_bf16(a, b, ghz);
        b.q[0] = *(const u32x4*)pn; b.q[1] = *(const u32x4*)(pn + 16);
        ghn = wmma_bf16(a, b, ghn);
    }

    const float bir = bih[ncol], biz = bih[DM + ncol], bin2 = bih[2 * DM + ncol];
    const float bhr = bhh[ncol], bhz = bhh[DM + ncol], bhn  = bhh[2 * DM + ncol];
#pragma unroll
    for (int r = 0; r < 8; ++r) {
        int m = hl * 8 + r;
        float rg = sigmf(gir[r] + bir + ghr[r] + bhr);
        float zg = sigmf(giz[r] + biz + ghz[r] + bhz);
        float ng = tanhf(gin[r] + bin2 + rg * (ghn[r] + bhn));
        float h  = hF[m][ncol];
        float nm = (1.f - zg) * ng + zg * h;
        if (winS[m]) outMem[(size_t)rowS[m] * DM + ncol] = nm;
    }
}

// ------------------------------------------------------------------
// embeddings = [updMem | nodeF] @ We^T + be ; 16 nodes/block, 8 waves x 16 cols
__global__ __launch_bounds__(256) void tgn_embed_kernel(
    const float* __restrict__ updMem, const float* __restrict__ nodeF,
    const unsigned short* __restrict__ Webf, const float* __restrict__ be,
    float* __restrict__ emb)
{
    __shared__ __attribute__((aligned(16))) unsigned short aT[16][256];
    const int tid = threadIdx.x;
    const int n0  = blockIdx.x * 16;

    for (int idx = tid; idx < 16 * 256; idx += 256) {
        int m = idx >> 8, k = idx & 255;
        float v = (k < DM) ? updMem[(size_t)(n0 + m) * DM + k]
                           : nodeF[(size_t)(n0 + m) * DM + (k - DM)];
        aT[m][k] = f32_to_bf16(v);
    }
    __syncthreads();

    const int wv   = tid >> 5;
    const int lane = tid & 31;
    const int hl   = lane >> 4;
    const int mr   = lane & 15;
    const int koff = hl * 8;
    const int ncol = wv * 16 + mr;

    BF16Frag a, b;
    v8f acc = {0.f,0.f,0.f,0.f,0.f,0.f,0.f,0.f};
#pragma unroll
    for (int k0 = 0; k0 < 256; k0 += 32) {
        a.q[0] = *(const u32x4*)&aT[mr][k0 + koff];
        a.q[1] = *(const u32x4*)&aT[mr][k0 + 16 + koff];
        const unsigned short* bp = Webf + (size_t)ncol * 256 + k0 + koff;
        b.q[0] = *(const u32x4*)bp;
        b.q[1] = *(const u32x4*)(bp + 16);
        acc = wmma_bf16(a, b, acc);
    }
    const float bias = be[ncol];
#pragma unroll
    for (int r = 0; r < 8; ++r)
        emb[(size_t)(n0 + hl * 8 + r) * DM + ncol] = acc[r] + bias;
}

// ------------------------------------------------------------------
extern "C" void kernel_launch(void* const* d_in, const int* in_sizes, int n_in,
                              void* d_out, int out_size, void* d_ws, size_t ws_size,
                              hipStream_t stream) {
    const float* nodeF  = (const float*)d_in[0];
    const int*   eidx   = (const int*)d_in[1];
    const float* efeat  = (const float*)d_in[2];
    const float* memory = (const float*)d_in[3];
    const float* W1  = (const float*)d_in[4];  const float* b1  = (const float*)d_in[5];
    const float* W2  = (const float*)d_in[6];  const float* b2  = (const float*)d_in[7];
    const float* Wih = (const float*)d_in[8];  const float* bih = (const float*)d_in[9];
    const float* Whh = (const float*)d_in[10]; const float* bhh = (const float*)d_in[11];
    const float* We  = (const float*)d_in[12]; const float* be  = (const float*)d_in[13];

    float* emb    = (float*)d_out;
    float* updMem = (float*)d_out + (size_t)N_NODES * DM;

    unsigned short* wbf = (unsigned short*)d_ws;
    int* winner = (int*)((char*)d_ws + OFF_WINNER_BYTES);

    tgn_convert_weights<<<(TOT_BF + 255) / 256, 256, 0, stream>>>(W1, W2, Wih, Whh, We, wbf);
    tgn_init<<<((unsigned)N_NODES * DM + 255) / 256, 256, 0, stream>>>(memory, updMem, winner);
    tgn_winner<<<(N_EDGES + 255) / 256, 256, 0, stream>>>(eidx, winner);
    tgn_edge_kernel<<<N_EDGES / 16, 256, 0, stream>>>(
        eidx, efeat, memory,
        wbf + OFF_W1, b1, wbf + OFF_W2, b2,
        wbf + OFF_WIH, bih, wbf + OFF_WHH, bhh,
        winner, updMem);
    tgn_embed_kernel<<<N_NODES / 16, 256, 0, stream>>>(updMem, nodeF, wbf + OFF_WE, be, emb);
}
